// NanoLlama_36730560315820
// MI455X (gfx1250) — compile-verified, matrix operand order checked by reading, not run on hardware
//
#include <hip/hip_runtime.h>
#include <hip/hip_bf16.h>
#include <stdint.h>

#define DIMD    1024
#define NLAYERS 4
#define NHEADS  16
#define HDIM    64
#define HIDDEN  4096
#define NB      2
#define NT      1024
#define MTOT    (NB*NT)     // 2048 rows
#define NVOCAB  32000

typedef __bf16 bf16;
typedef __attribute__((ext_vector_type(16))) __bf16 v16bf;
typedef __attribute__((ext_vector_type(8)))  __bf16 v8bf;
typedef __attribute__((ext_vector_type(8)))  float  v8f;
typedef __attribute__((ext_vector_type(4)))  float  v4f;

union FragA { v16bf v; v8bf h[2]; };

__device__ __forceinline__ bf16 f2bf(float f) {
  union { float f; uint32_t u; } x; x.f = f;
  uint32_t r = x.u + 0x7FFFu + ((x.u >> 16) & 1u);   // round-to-nearest-even
  uint16_t hv = (uint16_t)(r >> 16);
  return __builtin_bit_cast(bf16, hv);
}

// ---------------- fp32 -> bf16 weight conversion ----------------
__global__ __launch_bounds__(256) void k_f32_to_bf16(const float* __restrict__ s,
                                                     bf16* __restrict__ d, int n) {
  int i = (blockIdx.x * 256 + threadIdx.x) * 4;
  if (i + 3 < n) {
    v4f v = *(const v4f*)(s + i);
    d[i + 0] = f2bf(v.x); d[i + 1] = f2bf(v.y);
    d[i + 2] = f2bf(v.z); d[i + 3] = f2bf(v.w);
  } else {
    for (; i < n; ++i) d[i] = f2bf(s[i]);
  }
}

// ---------------- embedding gather ----------------
__global__ __launch_bounds__(256) void k_embed(const int* __restrict__ tok,
                                               const float* __restrict__ emb,
                                               float* __restrict__ h) {
  int idx = blockIdx.x * 256 + threadIdx.x;      // MTOT*DIMD/4 threads
  int row = idx >> 8;                            // 256 vec4 per row
  int c4  = (idx & 255) * 4;
  int t = tok[row];
  *(v4f*)&h[(size_t)row * DIMD + c4] = *(const v4f*)&emb[(size_t)t * DIMD + c4];
}

// ---------------- RMSNorm -> bf16 ----------------
__global__ __launch_bounds__(256) void k_rmsnorm_bf16(const float* __restrict__ x,
                                                      const float* __restrict__ w,
                                                      bf16* __restrict__ o) {
  __shared__ float red[8];
  int row = blockIdx.x, tid = threadIdx.x;
  const float* xr = x + (size_t)row * DIMD;
  v4f v = *(const v4f*)&xr[tid * 4];
  float ss = v.x*v.x + v.y*v.y + v.z*v.z + v.w*v.w;
  #pragma unroll
  for (int m = 16; m >= 1; m >>= 1) ss += __shfl_xor(ss, m, 32);
  if ((tid & 31) == 0) red[tid >> 5] = ss;
  __syncthreads();
  float tot = 0.f;
  #pragma unroll
  for (int i = 0; i < 8; ++i) tot += red[i];
  float sc = rsqrtf(tot * (1.0f / DIMD) + 1e-6f);
  v4f wv = *(const v4f*)&w[tid * 4];
  bf16* op = o + (size_t)row * DIMD + tid * 4;
  op[0] = f2bf(v.x * wv.x * sc); op[1] = f2bf(v.y * wv.y * sc);
  op[2] = f2bf(v.z * wv.z * sc); op[3] = f2bf(v.w * wv.w * sc);
}

// ---------------- tiled bf16 WMMA GEMM:  C = A[M,K] * B[N,K]^T ----------------
// Software-pipelined: global loads for tile kt+32 are issued before the WMMA
// work on tile kt, hiding HBM/L2 latency behind the matrix pipe.
// MODE 0: Cf = acc*oscale                      (lm_head -> d_out, fp32)
// MODE 1: Cf = res + acc                       (residual add, fp32, in-place safe)
// MODE 2: Cb[(b*H+head)*T + t][dh] = bf16(acc*oscale)   (QKV scatter)
// MODE 3: Cb = bf16(silu(acc)*acc2)            (fused gate/up, B1 = Wu)
template <int MODE>
__global__ __launch_bounds__(256) void k_gemm(
    const bf16* __restrict__ A, const bf16* __restrict__ B0,
    const bf16* __restrict__ B1, const float* __restrict__ res,
    float* __restrict__ Cf, bf16* __restrict__ Cb,
    int M, int N, int K, float oscale) {
  __shared__ bf16 As[128][40];                      // 32-col K tile, +8 pad (16B aligned rows)
  __shared__ bf16 Bs[64][40];
  __shared__ bf16 Bs2[(MODE == 3) ? 64 : 1][40];
  (void)M;
  int tid = threadIdx.x;
  int lane = tid & 31, wid = tid >> 5;
  int lr = lane & 15, lh = lane >> 4;
  int wm = wid >> 1, wn = wid & 1;                  // 4x2 wave grid -> 128x64 block tile
  size_t rowA0 = (size_t)blockIdx.x * 128;
  size_t colB0 = (size_t)blockIdx.y * 64;
  v8f acc[2][2] = {}; v8f acc2[2][2] = {};

  // register staging for the software pipeline
  v8bf areg[2]; v8bf breg; v8bf b2reg;
  const int rA = tid >> 2,  qA = (tid & 3) * 8;     // A chunk 0 coords (c = tid)
  const int rA2 = (tid + 256) >> 2, qA2 = ((tid + 256) & 3) * 8;
  const int rB = tid >> 2,  qB = (tid & 3) * 8;

  auto loadTile = [&](int kt) {
    areg[0] = *(const v8bf*)&A[(rowA0 + rA)  * K + kt + qA];
    areg[1] = *(const v8bf*)&A[(rowA0 + rA2) * K + kt + qA2];
    breg    = *(const v8bf*)&B0[(colB0 + rB) * K + kt + qB];
    if constexpr (MODE == 3)
      b2reg = *(const v8bf*)&B1[(colB0 + rB) * K + kt + qB];
  };
  auto storeTile = [&]() {
    *(v8bf*)&As[rA][qA]   = areg[0];
    *(v8bf*)&As[rA2][qA2] = areg[1];
    *(v8bf*)&Bs[rB][qB]   = breg;
    if constexpr (MODE == 3) *(v8bf*)&Bs2[rB][qB] = b2reg;
  };

  loadTile(0);
  storeTile();
  for (int kt = 0; kt < K; kt += 32) {
    __syncthreads();                                 // LDS tile visible
    bool more = (kt + 32) < K;
    if (more) loadTile(kt + 32);                     // overlap with WMMA below

    FragA a[2], b[2], b2[2];
    #pragma unroll
    for (int i = 0; i < 2; ++i) {
      int r = wm * 32 + i * 16 + lr;
      a[i].h[0] = *(const v8bf*)&As[r][lh * 8];
      a[i].h[1] = *(const v8bf*)&As[r][16 + lh * 8];
    }
    #pragma unroll
    for (int j = 0; j < 2; ++j) {
      int r = wn * 32 + j * 16 + lr;
      b[j].h[0] = *(const v8bf*)&Bs[r][lh * 8];
      b[j].h[1] = *(const v8bf*)&Bs[r][16 + lh * 8];
      if constexpr (MODE == 3) {
        b2[j].h[0] = *(const v8bf*)&Bs2[r][lh * 8];
        b2[j].h[1] = *(const v8bf*)&Bs2[r][16 + lh * 8];
      }
    }
    #pragma unroll
    for (int i = 0; i < 2; ++i)
      #pragma unroll
      for (int j = 0; j < 2; ++j) {
        acc[i][j] = __builtin_amdgcn_wmma_f32_16x16x32_bf16(
            false, a[i].v, false, b[j].v, (short)0, acc[i][j], false, false);
        if constexpr (MODE == 3)
          acc2[i][j] = __builtin_amdgcn_wmma_f32_16x16x32_bf16(
              false, a[i].v, false, b2[j].v, (short)0, acc2[i][j], false, false);
      }

    __syncthreads();                                 // all waves done reading
    if (more) storeTile();
  }

  #pragma unroll
  for (int i = 0; i < 2; ++i)
    #pragma unroll
    for (int j = 0; j < 2; ++j)
      #pragma unroll
      for (int r = 0; r < 8; ++r) {
        size_t gm = rowA0 + wm * 32 + i * 16 + lh * 8 + r;   // C layout: lane holds rows lh*8+r
        size_t gn = colB0 + wn * 32 + j * 16 + lr;           // col = lane%16
        float va = acc[i][j][r];
        if constexpr (MODE == 0) {
          Cf[gm * N + gn] = va * oscale;
        } else if constexpr (MODE == 1) {
          Cf[gm * N + gn] = res[gm * N + gn] + va;
        } else if constexpr (MODE == 2) {
          int bI = (int)(gm >> 10), t = (int)(gm & 1023);
          int hd = (int)(gn >> 6),  dh = (int)(gn & 63);
          Cb[(((size_t)(bI * NHEADS + hd)) * NT + t) * HDIM + dh] = f2bf(va * oscale);
        } else {
          float g = va, u = acc2[i][j][r];
          float sg = g / (1.f + __expf(-g));
          Cb[gm * N + gn] = f2bf(sg * u);
        }
      }
}

// ---------------- flash attention (no mask), WMMA, online softmax ----------------
// grid: (T/64, B*H), block: 128 (4 waves x 16 q-rows). q is pre-scaled by 1/sqrt(HDIM).
// K/V tiles are software-pipelined like the GEMM.
__global__ __launch_bounds__(128) void k_attn(const bf16* __restrict__ q,
                                              const bf16* __restrict__ k,
                                              const bf16* __restrict__ v,
                                              bf16* __restrict__ y) {
  __shared__ bf16 Ks[64][72];        // K tile [kv][dh], +8 pad
  __shared__ bf16 Vt[64][72];        // V^T tile [dh][kv]
  __shared__ bf16 Ps[4][16][72];     // per-wave P tile [qrow][kv]
  int tid = threadIdx.x;
  int lane = tid & 31, wid = tid >> 5;
  int lr = lane & 15, lh = lane >> 4;
  int bh = blockIdx.y;
  int bI = bh >> 4, hd = bh & 15;
  size_t base = (size_t)bh * NT * HDIM;
  int q0 = blockIdx.x * 64 + wid * 16;

  FragA qa[2];
  { const bf16* qr = q + base + (size_t)(q0 + lr) * HDIM;    // A-frag: lane = row lr
    qa[0].h[0] = *(const v8bf*)&qr[lh * 8];
    qa[0].h[1] = *(const v8bf*)&qr[16 + lh * 8];
    qa[1].h[0] = *(const v8bf*)&qr[32 + lh * 8];
    qa[1].h[1] = *(const v8bf*)&qr[48 + lh * 8]; }

  float mrow[8], lrow[8]; v8f yacc[4] = {};
  #pragma unroll
  for (int r = 0; r < 8; ++r) { mrow[r] = -1e30f; lrow[r] = 0.f; }

  // register staging for pipelined K/V tiles: 4 chunks each
  v8bf kreg[4], vreg[4];
  auto loadKV = [&](int kvb) {
    #pragma unroll
    for (int i = 0; i < 4; ++i) {
      int c = tid + i * 128;
      int r = c >> 3, col = (c & 7) * 8;
      kreg[i] = *(const v8bf*)&k[base + (size_t)(kvb + r) * HDIM + col];
      vreg[i] = *(const v8bf*)&v[base + (size_t)(kvb + r) * HDIM + col];
    }
  };
  auto storeKV = [&]() {
    #pragma unroll
    for (int i = 0; i < 4; ++i) {
      int c = tid + i * 128;
      int r = c >> 3, col = (c & 7) * 8;
      *(v8bf*)&Ks[r][col] = kreg[i];
      #pragma unroll
      for (int e = 0; e < 8; ++e) Vt[col + e][r] = vreg[i][e];   // transpose V
    }
  };

  loadKV(0);
  storeKV();
  for (int kvb = 0; kvb < NT; kvb += 64) {
    __syncthreads();                          // K/V tile visible
    bool more = (kvb + 64) < NT;
    if (more) loadKV(kvb + 64);               // overlap with compute below

    v8f s[4];
    #pragma unroll
    for (int j = 0; j < 4; ++j) {             // S tile 16x64 via 4x(2 wmma)
      FragA kb0, kb1;
      int r = j * 16 + lr;
      kb0.h[0] = *(const v8bf*)&Ks[r][lh * 8];
      kb0.h[1] = *(const v8bf*)&Ks[r][16 + lh * 8];
      kb1.h[0] = *(const v8bf*)&Ks[r][32 + lh * 8];
      kb1.h[1] = *(const v8bf*)&Ks[r][48 + lh * 8];
      v8f z = {};
      z = __builtin_amdgcn_wmma_f32_16x16x32_bf16(false, qa[0].v, false, kb0.v, (short)0, z, false, false);
      z = __builtin_amdgcn_wmma_f32_16x16x32_bf16(false, qa[1].v, false, kb1.v, (short)0, z, false, false);
      s[j] = z;
    }

    float mnew[8], alpha[8], psum[8];
    #pragma unroll
    for (int r = 0; r < 8; ++r) {
      float mx = fmaxf(fmaxf(s[0][r], s[1][r]), fmaxf(s[2][r], s[3][r]));
      #pragma unroll
      for (int msk = 1; msk < 16; msk <<= 1) mx = fmaxf(mx, __shfl_xor(mx, msk, 32));
      mnew[r] = fmaxf(mrow[r], mx);
      alpha[r] = __expf(mrow[r] - mnew[r]);
      lrow[r] *= alpha[r];
      psum[r] = 0.f;
    }
    #pragma unroll
    for (int j = 0; j < 4; ++j)
      #pragma unroll
      for (int r = 0; r < 8; ++r) {
        float p = __expf(s[j][r] - mnew[r]);
        psum[r] += p;
        yacc[j][r] *= alpha[r];
        Ps[wid][lh * 8 + r][j * 16 + lr] = f2bf(p);   // C-layout -> LDS
      }
    #pragma unroll
    for (int r = 0; r < 8; ++r) {
      float t = psum[r];
      #pragma unroll
      for (int msk = 1; msk < 16; msk <<= 1) t += __shfl_xor(t, msk, 32);
      lrow[r] += t;
      mrow[r] = mnew[r];
    }

    FragA pa[2];                                       // reload P as A-frags
    pa[0].h[0] = *(const v8bf*)&Ps[wid][lr][lh * 8];
    pa[0].h[1] = *(const v8bf*)&Ps[wid][lr][16 + lh * 8];
    pa[1].h[0] = *(const v8bf*)&Ps[wid][lr][32 + lh * 8];
    pa[1].h[1] = *(const v8bf*)&Ps[wid][lr][48 + lh * 8];
    #pragma unroll
    for (int j = 0; j < 4; ++j) {
      FragA vb0, vb1;
      int r = j * 16 + lr;
      vb0.h[0] = *(const v8bf*)&Vt[r][lh * 8];
      vb0.h[1] = *(const v8bf*)&Vt[r][16 + lh * 8];
      vb1.h[0] = *(const v8bf*)&Vt[r][32 + lh * 8];
      vb1.h[1] = *(const v8bf*)&Vt[r][48 + lh * 8];
      yacc[j] = __builtin_amdgcn_wmma_f32_16x16x32_bf16(false, pa[0].v, false, vb0.v, (short)0, yacc[j], false, false);
      yacc[j] = __builtin_amdgcn_wmma_f32_16x16x32_bf16(false, pa[1].v, false, vb1.v, (short)0, yacc[j], false, false);
    }

    __syncthreads();                          // all waves done reading Ks/Vt
    if (more) storeKV();
  }

  #pragma unroll
  for (int j = 0; j < 4; ++j)
    #pragma unroll
    for (int r = 0; r < 8; ++r) {
      int t = blockIdx.x * 64 + wid * 16 + lh * 8 + r;
      int col = hd * HDIM + j * 16 + lr;
      float val = yacc[j][r] / lrow[r];
      y[((size_t)bI * NT + t) * DIMD + col] = f2bf(val);
    }
}

// ---------------- host orchestration ----------------
extern "C" void kernel_launch(void* const* d_in, const int* in_sizes, int n_in,
                              void* d_out, int out_size, void* d_ws, size_t ws_size,
                              hipStream_t stream) {
  (void)in_sizes; (void)n_in; (void)out_size; (void)ws_size;
  const int*   tokens = (const int*)  d_in[0];
  const float* embed  = (const float*)d_in[1];
  const float* Wq  = (const float*)d_in[2];
  const float* Wk  = (const float*)d_in[3];
  const float* Wv_ = (const float*)d_in[4];
  const float* Wo  = (const float*)d_in[5];
  const float* Wg  = (const float*)d_in[6];
  const float* Wu  = (const float*)d_in[7];
  const float* Wd  = (const float*)d_in[8];
  const float* ln1 = (const float*)d_in[9];
  const float* ln2 = (const float*)d_in[10];
  const float* nw  = (const float*)d_in[11];
  const float* lm  = (const float*)d_in[12];
  float* out = (float*)d_out;

  char* p = (char*)d_ws;
  auto wsalloc = [&](size_t bytes) -> char* {
    char* r = p; p += (bytes + 255) & ~(size_t)255; return r;
  };
  const size_t nQKVO = (size_t)NLAYERS * DIMD * DIMD;       // 4.19M each
  const size_t nFFN  = (size_t)NLAYERS * HIDDEN * DIMD;     // 16.8M each
  const size_t nLM   = (size_t)NVOCAB * DIMD;               // 32.8M
  bf16* WqB = (bf16*)wsalloc(nQKVO * 2);
  bf16* WkB = (bf16*)wsalloc(nQKVO * 2);
  bf16* WvB = (bf16*)wsalloc(nQKVO * 2);
  bf16* WoB = (bf16*)wsalloc(nQKVO * 2);
  bf16* WgB = (bf16*)wsalloc(nFFN * 2);
  bf16* WuB = (bf16*)wsalloc(nFFN * 2);
  bf16* WdB = (bf16*)wsalloc(nFFN * 2);
  bf16* lmB = (bf16*)wsalloc(nLM * 2);
  float* h  = (float*)wsalloc((size_t)MTOT * DIMD * 4);
  bf16* xb  = (bf16*)wsalloc((size_t)MTOT * DIMD * 2);
  bf16* qb  = (bf16*)wsalloc((size_t)MTOT * DIMD * 2);
  bf16* kb  = (bf16*)wsalloc((size_t)MTOT * DIMD * 2);
  bf16* vb  = (bf16*)wsalloc((size_t)MTOT * DIMD * 2);
  bf16* yb  = (bf16*)wsalloc((size_t)MTOT * DIMD * 2);
  bf16* mb  = (bf16*)wsalloc((size_t)MTOT * HIDDEN * 2);

  auto conv = [&](const float* s, bf16* d, size_t n) {
    int blocks = (int)((n / 4 + 255) / 256);
    k_f32_to_bf16<<<blocks, 256, 0, stream>>>(s, d, (int)n);
  };
  conv(Wq, WqB, nQKVO); conv(Wk, WkB, nQKVO); conv(Wv_, WvB, nQKVO); conv(Wo, WoB, nQKVO);
  conv(Wg, WgB, nFFN);  conv(Wu, WuB, nFFN);  conv(Wd, WdB, nFFN);
  conv(lm, lmB, nLM);

  k_embed<<<(MTOT * DIMD / 4) / 256, 256, 0, stream>>>(tokens, embed, h);

  dim3 gD(MTOT / 128, DIMD / 64);
  dim3 gH(MTOT / 128, HIDDEN / 64);
  dim3 gA(NT / 64, NB * NHEADS);
  const float qscale = 0.125f;    // 1/sqrt(64)

  for (int L = 0; L < NLAYERS; ++L) {
    const bf16* wq = WqB + (size_t)L * DIMD * DIMD;
    const bf16* wk = WkB + (size_t)L * DIMD * DIMD;
    const bf16* wv = WvB + (size_t)L * DIMD * DIMD;
    const bf16* wo = WoB + (size_t)L * DIMD * DIMD;
    const bf16* wg = WgB + (size_t)L * HIDDEN * DIMD;
    const bf16* wu = WuB + (size_t)L * HIDDEN * DIMD;
    const bf16* wd = WdB + (size_t)L * HIDDEN * DIMD;

    k_rmsnorm_bf16<<<MTOT, 256, 0, stream>>>(h, ln1 + (size_t)L * DIMD, xb);
    k_gemm<2><<<gD, 256, 0, stream>>>(xb, wq, nullptr, nullptr, nullptr, qb, MTOT, DIMD, DIMD, qscale);
    k_gemm<2><<<gD, 256, 0, stream>>>(xb, wk, nullptr, nullptr, nullptr, kb, MTOT, DIMD, DIMD, 1.f);
    k_gemm<2><<<gD, 256, 0, stream>>>(xb, wv, nullptr, nullptr, nullptr, vb, MTOT, DIMD, DIMD, 1.f);
    k_attn<<<gA, 128, 0, stream>>>(qb, kb, vb, yb);
    k_gemm<1><<<gD, 256, 0, stream>>>(yb, wo, nullptr, h, h, nullptr, MTOT, DIMD, DIMD, 1.f);
    k_rmsnorm_bf16<<<MTOT, 256, 0, stream>>>(h, ln2 + (size_t)L * DIMD, xb);
    k_gemm<3><<<gH, 256, 0, stream>>>(xb, wg, wu, nullptr, nullptr, mb, MTOT, HIDDEN, DIMD, 1.f);
    k_gemm<1><<<gD, 256, 0, stream>>>(mb, wd, nullptr, h, h, nullptr, MTOT, DIMD, HIDDEN, 1.f);
  }

  k_rmsnorm_bf16<<<MTOT, 256, 0, stream>>>(h, nw, xb);
  dim3 gV(MTOT / 128, NVOCAB / 64);
  k_gemm<0><<<gV, 256, 0, stream>>>(xb, lmB, nullptr, nullptr, out, nullptr, MTOT, NVOCAB, DIMD, 1.f);
}
